// MixHopNetwork_75299366633514
// MI455X (gfx1250) — compile-verified, matrix operand order checked by reading, not run on hardware
//
#include <hip/hip_runtime.h>
#include <stdint.h>

// ---------------------------------------------------------------------------
// MixHop forward for MI455X (gfx1250, wave32).
// Dense GEMMs: v_wmma_f32_16x16x32_bf16 with 2x2 register tiling per wave
// (4 WMMAs per K-step from 4 fragment loads -> matrix-pipe bound, not load
// bound). SpMM hops: f32 global atomics (operands L2-resident, 80MB << 192MB).
// ---------------------------------------------------------------------------

typedef __attribute__((ext_vector_type(16))) __bf16 v16bf;
typedef __attribute__((ext_vector_type(8)))  __bf16 v8bf;
typedef __attribute__((ext_vector_type(8)))  float  v8f;
typedef __attribute__((ext_vector_type(4)))  float  v4f;

#define NNODES 100000
#define NEDGES 3200000
#define FIN    512
#define HID    200
#define NCLS   40
#define COLS   608   // 600 padded up to 19*32 (K and N padding for WMMA tiles)
#define C3PAD  64    // 40 classes padded to two 32-wide wave tiles

// ------------------------------ utility kernels ----------------------------

__global__ void zero_u32_kernel(uint32_t* __restrict__ p, long n) {
    long i = (long)blockIdx.x * blockDim.x + threadIdx.x;
    if (i < n) p[i] = 0u;
}

// Wu [3,512,200] f32 -> W1T bf16, column-major per output col: W1T[c*512 + k]
__global__ void pack_wu_kernel(const float* __restrict__ Wu, __bf16* __restrict__ W1T) {
    int idx = blockIdx.x * 256 + threadIdx.x;
    if (idx >= COLS * FIN) return;
    int c = idx / FIN, k = idx % FIN;
    float v = 0.0f;
    if (c < 600) { int i = c / HID, j = c % HID; v = Wu[((long)i * FIN + k) * HID + j]; }
    W1T[idx] = (__bf16)v;
}

// Wb [3,600,200] f32 -> W2T bf16 [c*608 + k], zero-padded K rows 600..607
__global__ void pack_wb_kernel(const float* __restrict__ Wb, __bf16* __restrict__ W2T) {
    int idx = blockIdx.x * 256 + threadIdx.x;
    if (idx >= COLS * COLS) return;
    int c = idx / COLS, k = idx % COLS;
    float v = 0.0f;
    if (c < 600 && k < 600) { int i = c / HID, j = c % HID; v = Wb[((long)i * 600 + k) * HID + j]; }
    W2T[idx] = (__bf16)v;
}

// fc_w [600,40] f32 -> W3T bf16 [c*608 + k], c padded to 64, k padded to 608
__global__ void pack_fc_kernel(const float* __restrict__ fcw, __bf16* __restrict__ W3T) {
    int idx = blockIdx.x * 256 + threadIdx.x;
    if (idx >= C3PAD * COLS) return;
    int c = idx / COLS, k = idx % COLS;
    float v = 0.0f;
    if (c < NCLS && k < 600) v = fcw[(long)k * NCLS + c];
    W3T[idx] = (__bf16)v;
}

__global__ void pack_bias_kernel(const float* __restrict__ bu, const float* __restrict__ fcb,
                                 float* __restrict__ b1, float* __restrict__ b3) {
    int i = blockIdx.x * 256 + threadIdx.x;
    if (i < COLS)  b1[i] = (i < 600)  ? bu[i]  : 0.0f;
    if (i < C3PAD) b3[i] = (i < NCLS) ? fcb[i] : 0.0f;
}

// ---------------------- WMMA fragment load helpers -------------------------
// Per ISA 7.12.2 (wave32, 16-bit A 16x32): lane<16 holds K 0-7 / 16-23,
// lane>=16 holds K 8-15 / 24-31; B mirrored per output column. `base` points
// at element (row-or-col, k0 + khalf); elements 8..15 sit 16 elements later.

__device__ __forceinline__ v16bf load_frag_bf16(const __bf16* __restrict__ base) {
    v8bf lo = *(const v8bf*)(base + 0);
    v8bf hi = *(const v8bf*)(base + 16);
    v16bf r;
#pragma unroll
    for (int m = 0; m < 8; ++m) { r[m] = lo[m]; r[8 + m] = hi[m]; }
    return r;
}

__device__ __forceinline__ v16bf load_frag_f32(const float* __restrict__ base) {
    v4f f0 = *(const v4f*)(base + 0);
    v4f f1 = *(const v4f*)(base + 4);
    v4f f2 = *(const v4f*)(base + 16);
    v4f f3 = *(const v4f*)(base + 20);
    v16bf r;
#pragma unroll
    for (int m = 0; m < 4; ++m) {
        r[m]      = (__bf16)f0[m];
        r[4 + m]  = (__bf16)f1[m];
        r[8 + m]  = (__bf16)f2[m];
        r[12 + m] = (__bf16)f3[m];
    }
    return r;
}

// ------------------------------ WMMA GEMM ----------------------------------
// C[M, NC] = op(A[M,K] @ B[K,NC]); B packed bf16, K-contiguous per column
// (BT[c*ldb + k]). Wave tile = 32x32 (2x2 fragments, 4 WMMA per K-step).
// Block = 4 waves covering 32 rows x 128 cols. M must be a multiple of 32.
template<bool A_IS_F32, bool RELU, bool HAS_BIAS>
__global__ __launch_bounds__(128)
void wmma_gemm_kernel(const void* __restrict__ Av, int lda,
                      const __bf16* __restrict__ BT, int ldb,
                      const float* __restrict__ bias,
                      float* __restrict__ C, int ldc,
                      int n_col_tiles32, int ksteps) {
    const int lane = threadIdx.x & 31;
    const int wave = threadIdx.x >> 5;
    const int ct = blockIdx.x * 4 + wave;         // 32-wide column tile index
    if (ct >= n_col_tiles32) return;              // wave-uniform exit (no barriers)

    const int row0  = blockIdx.y * 32;
    const int col0  = ct * 32;
    const int half  = lane >> 4;                  // 0: K 0-7/16-23, 1: K 8-15/24-31
    const int l16   = lane & 15;
    const int khalf = half * 8;

    const int arow = row0 + l16;                  // + 16 for second row fragment
    const int bcol = col0 + l16;                  // + 16 for second col fragment

    const __bf16* __restrict__ Ab = (const __bf16*)Av;
    const float*  __restrict__ Af = (const float*)Av;

    v8f acc00 = {}, acc01 = {}, acc10 = {}, acc11 = {};

    for (int ks = 0; ks < ksteps; ++ks) {
        const int k0 = ks * 32;
        v16bf a0, a1;

        if (A_IS_F32) {
            const float* p0 = Af + (long)arow * lda + k0 + khalf;
            const float* p1 = p0 + (long)16 * lda;
            a0 = load_frag_f32(p0);
            a1 = load_frag_f32(p1);
            __builtin_prefetch(p0 + 32, 0, 1);    // speculative; OOB silently dropped
            __builtin_prefetch(p1 + 32, 0, 1);
        } else {
            const __bf16* p0 = Ab + (long)arow * lda + k0 + khalf;
            const __bf16* p1 = p0 + (long)16 * lda;
            a0 = load_frag_bf16(p0);
            a1 = load_frag_bf16(p1);
            __builtin_prefetch(p0 + 32, 0, 1);
            __builtin_prefetch(p1 + 32, 0, 1);
        }
        const __bf16* q0 = BT + (long)bcol * ldb + k0 + khalf;
        v16bf b0 = load_frag_bf16(q0);
        v16bf b1 = load_frag_bf16(q0 + (long)16 * ldb);

        // 4 independent WMMAs per K-step (no D->A/B hazards between them)
        acc00 = __builtin_amdgcn_wmma_f32_16x16x32_bf16(false, a0, false, b0, (short)0, acc00, false, false);
        acc01 = __builtin_amdgcn_wmma_f32_16x16x32_bf16(false, a0, false, b1, (short)0, acc01, false, false);
        acc10 = __builtin_amdgcn_wmma_f32_16x16x32_bf16(false, a1, false, b0, (short)0, acc10, false, false);
        acc11 = __builtin_amdgcn_wmma_f32_16x16x32_bf16(false, a1, false, b1, (short)0, acc11, false, false);
    }

    // Epilogue. C/D layout: row = v + 8*half, col = l16 within each 16x16 tile.
    const float bv0 = HAS_BIAS ? bias[col0 + l16] : 0.0f;
    const float bv1 = HAS_BIAS ? bias[col0 + 16 + l16] : 0.0f;
    const v8f* accs[4] = { &acc00, &acc01, &acc10, &acc11 };
#pragma unroll
    for (int t = 0; t < 4; ++t) {
        const int ti = t >> 1, tj = t & 1;        // row / col sub-tile
        const long rbase = row0 + 16 * ti + 8 * half;
        const int  c     = col0 + 16 * tj + l16;
        const float bv   = tj ? bv1 : bv0;
        const v8f av = *accs[t];
#pragma unroll
        for (int v = 0; v < 8; ++v) {
            float x = av[v] + bv;
            if (RELU) x = fmaxf(x, 0.0f);
            C[(rbase + v) * ldc + c] = x;
        }
    }
}

// ------------------------------ SpMM (COO) ---------------------------------
// y[row] += val * x[col]; x strided slice, y compact [N,200]. 64 threads/edge.
__global__ __launch_bounds__(256)
void spmm_kernel(const int* __restrict__ row, const int* __restrict__ col,
                 const float* __restrict__ val,
                 const float* __restrict__ x, int xstride,
                 float* __restrict__ y, int E) {
    const int t  = threadIdx.x;
    const int e  = blockIdx.x * 4 + (t >> 6);
    const int f4 = t & 63;
    if (e >= E || f4 >= HID / 4) return;
    const int r = row[e], c = col[e];
    const float v = val[e];
    v4f xv = *(const v4f*)(x + (long)c * xstride + f4 * 4);
    float* yp = y + (long)r * HID + f4 * 4;
    atomicAdd(yp + 0, v * xv[0]);
    atomicAdd(yp + 1, v * xv[1]);
    atomicAdd(yp + 2, v * xv[2]);
    atomicAdd(yp + 3, v * xv[3]);
}

// f32 [N, sstride] slice (+optional bias) -> bf16 into ab[N,608] at column dstoff
__global__ void cvt_slice_kernel(const float* __restrict__ src, int sstride,
                                 __bf16* __restrict__ dst, int dstoff,
                                 const float* __restrict__ bias) {
    long idx = (long)blockIdx.x * 256 + threadIdx.x;
    if (idx >= (long)NNODES * HID) return;
    int n = (int)(idx / HID), j = (int)(idx % HID);
    float v = src[(long)n * sstride + j];
    if (bias) v += bias[j];
    dst[(long)n * COLS + dstoff + j] = (__bf16)v;
}

// ------------------------------ log-softmax --------------------------------
__global__ void logsoftmax_kernel(const float* __restrict__ logits, float* __restrict__ out) {
    int n = blockIdx.x * 256 + threadIdx.x;
    if (n >= NNODES) return;
    const float* p = logits + (long)n * C3PAD;
    float m = -3.4e38f;
#pragma unroll
    for (int j = 0; j < NCLS; ++j) m = fmaxf(m, p[j]);
    float s = 0.0f;
#pragma unroll
    for (int j = 0; j < NCLS; ++j) s += __expf(p[j] - m);
    float ls = __logf(s);
#pragma unroll
    for (int j = 0; j < NCLS; ++j) out[(long)n * NCLS + j] = p[j] - m - ls;
}

// ------------------------------ host driver --------------------------------

extern "C" void kernel_launch(void* const* d_in, const int* in_sizes, int n_in,
                              void* d_out, int out_size, void* d_ws, size_t ws_size,
                              hipStream_t stream) {
    const float* features = (const float*)d_in[0];   // [N,512] f32
    const int*   adj_row  = (const int*)d_in[1];     // [E]
    const int*   adj_col  = (const int*)d_in[2];     // [E]
    const float* adj_val  = (const float*)d_in[3];   // [E]
    const float* Wu       = (const float*)d_in[4];   // [3,512,200]
    const float* bu       = (const float*)d_in[5];   // [3,200]
    const float* Wb       = (const float*)d_in[6];   // [3,600,200]
    const float* bb       = (const float*)d_in[7];   // [3,200]
    const float* fc_w     = (const float*)d_in[8];   // [600,40]
    const float* fc_b     = (const float*)d_in[9];   // [40]
    float* out = (float*)d_out;                      // [N,40]

    // bump allocator over workspace (256B aligned)
    char* wsp = (char*)d_ws;
    auto alloc = [&](size_t bytes) -> char* {
        char* p = wsp;
        wsp += (bytes + 255) & ~(size_t)255;
        return p;
    };
    __bf16* W1T    = (__bf16*)alloc((size_t)COLS * FIN * 2);     // packed Wu
    __bf16* W2T    = (__bf16*)alloc((size_t)COLS * COLS * 2);    // packed Wb
    __bf16* W3T    = (__bf16*)alloc((size_t)C3PAD * COLS * 2);   // packed fc_w
    float*  bias1  = (float*)alloc((size_t)COLS * 4);
    float*  bias3  = (float*)alloc((size_t)C3PAD * 4);
    __bf16* ab     = (__bf16*)alloc((size_t)NNODES * COLS * 2);  // abstract1 then abstract2, bf16
    float*  P      = (float*)alloc((size_t)NNODES * COLS * 4);   // relu(X@Wu) then abstract1@Wb
    float*  T      = (float*)alloc((size_t)NNODES * HID * 4);    // spmm temp
    float*  T2     = (float*)alloc((size_t)NNODES * HID * 4);    // spmm temp
    float*  logits = (float*)alloc((size_t)NNODES * C3PAD * 4);

    const int ROW_BLK = NNODES / 32;                   // 3125, exact
    const long NHID = (long)NNODES * HID;
    const dim3 gemm_big(5, ROW_BLK);                   // 19 col tiles (32-wide) / 4 waves
    const dim3 gemm_fc(1, ROW_BLK);                    // 2 col tiles (waves 0,1 active)
    const int spmm_blocks = (NEDGES + 3) / 4;

    auto zero_f32 = [&](float* p, long n) {
        zero_u32_kernel<<<(int)((n + 255) / 256), 256, 0, stream>>>((uint32_t*)p, n);
    };

    // 1) pack weights / biases to bf16, WMMA-friendly layouts
    pack_wu_kernel<<<(COLS * FIN + 255) / 256, 256, 0, stream>>>(Wu, W1T);
    pack_wb_kernel<<<(COLS * COLS + 255) / 256, 256, 0, stream>>>(Wb, W2T);
    pack_fc_kernel<<<(C3PAD * COLS + 255) / 256, 256, 0, stream>>>(fc_w, W3T);
    pack_bias_kernel<<<3, 256, 0, stream>>>(bu, fc_b, bias1, bias3);
    zero_u32_kernel<<<(int)(((long)NNODES * COLS * 2 / 4 + 255) / 256), 256, 0, stream>>>(
        (uint32_t*)ab, (long)NNODES * COLS * 2 / 4);   // zero incl. pad cols 600..607

    // 2) P = relu(X @ Wu_cat + bu_cat)   [N,608], pad cols compute to 0
    wmma_gemm_kernel<true, true, true><<<gemm_big, 128, 0, stream>>>(
        features, FIN, W1T, FIN, bias1, P, COLS, COLS / 32, FIN / 32);

    // 3) abstract1 (bf16): [ h0 | A@h1 | A@A@h2 ]
    cvt_slice_kernel<<<(int)((NHID + 255) / 256), 256, 0, stream>>>(P, COLS, ab, 0, nullptr);
    zero_f32(T, NHID);
    spmm_kernel<<<spmm_blocks, 256, 0, stream>>>(adj_row, adj_col, adj_val, P + HID, COLS, T, NEDGES);
    cvt_slice_kernel<<<(int)((NHID + 255) / 256), 256, 0, stream>>>(T, HID, ab, HID, nullptr);
    zero_f32(T2, NHID);
    spmm_kernel<<<spmm_blocks, 256, 0, stream>>>(adj_row, adj_col, adj_val, P + 2 * HID, COLS, T2, NEDGES);
    zero_f32(T, NHID);
    spmm_kernel<<<spmm_blocks, 256, 0, stream>>>(adj_row, adj_col, adj_val, T2, HID, T, NEDGES);
    cvt_slice_kernel<<<(int)((NHID + 255) / 256), 256, 0, stream>>>(T, HID, ab, 2 * HID, nullptr);

    // 4) P = abstract1 @ Wb_cat (bias applied after the hops)
    wmma_gemm_kernel<false, false, false><<<gemm_big, 128, 0, stream>>>(
        ab, COLS, W2T, COLS, nullptr, P, COLS, COLS / 32, COLS / 32);

    // 5) abstract2 (bf16): [ p0+bb0 | A@p1+bb1 | A@A@p2+bb2 ]
    cvt_slice_kernel<<<(int)((NHID + 255) / 256), 256, 0, stream>>>(P, COLS, ab, 0, bb);
    zero_f32(T, NHID);
    spmm_kernel<<<spmm_blocks, 256, 0, stream>>>(adj_row, adj_col, adj_val, P + HID, COLS, T, NEDGES);
    cvt_slice_kernel<<<(int)((NHID + 255) / 256), 256, 0, stream>>>(T, HID, ab, HID, bb + HID);
    zero_f32(T2, NHID);
    spmm_kernel<<<spmm_blocks, 256, 0, stream>>>(adj_row, adj_col, adj_val, P + 2 * HID, COLS, T2, NEDGES);
    zero_f32(T, NHID);
    spmm_kernel<<<spmm_blocks, 256, 0, stream>>>(adj_row, adj_col, adj_val, T2, HID, T, NEDGES);
    cvt_slice_kernel<<<(int)((NHID + 255) / 256), 256, 0, stream>>>(T, HID, ab, 2 * HID, bb + 2 * HID);

    // 6) logits = abstract2 @ fc_w + fc_b   [N,64] padded
    wmma_gemm_kernel<false, false, true><<<gemm_fc, 128, 0, stream>>>(
        ab, COLS, W3T, COLS, bias3, logits, C3PAD, C3PAD / 32, COLS / 32);

    // 7) row-wise log_softmax over the 40 valid classes
    logsoftmax_kernel<<<(NNODES + 255) / 256, 256, 0, stream>>>(logits, out);

    (void)in_sizes; (void)n_in; (void)out_size; (void)ws_size;
}
